// MMMambaEncoderLayer_120259084375
// MI455X (gfx1250) — matrix operand order repaired and checked by reading, not compiled, not run on hardware
//
#include <hip/hip_runtime.h>
#include <hip/hip_bf16.h>

typedef __attribute__((ext_vector_type(16))) _Float16 v16h;
typedef __attribute__((ext_vector_type(8)))  float    v8f;

// ---------------------------------------------------------------------------
// Pack a row-major f32 weight matrix B[K,N] into f16 WMMA B-fragment layout:
// Bp[((nt*kTiles + kt)*32 + lane)*16 + e] = B[kt*32 + (lane>=16?16:0) + e][nt*16 + lane%16]
// One thread per (nt, kt, lane), writing 16 contiguous halves (32 B).
// ---------------------------------------------------------------------------
__global__ void __launch_bounds__(256) pack_b_kernel(const float* __restrict__ B,
                                                     _Float16* __restrict__ Bp,
                                                     int N, int K, int ldb) {
  int idx = blockIdx.x * 256 + threadIdx.x;
  int kTiles = K >> 5;
  int total = (N >> 4) * kTiles * 32;
  if (idx >= total) return;
  int lane = idx & 31;
  int kt = (idx >> 5) % kTiles;
  int nt = (idx >> 5) / kTiles;
  int col  = nt * 16 + (lane & 15);
  int krow = kt * 32 + ((lane >> 4) << 4);
  const float* src = B + (size_t)krow * ldb + col;
  _Float16* dst = Bp + (size_t)idx * 16;
  #pragma unroll
  for (int e = 0; e < 16; e++) dst[e] = (_Float16)src[(size_t)e * ldb];
}

// f32 -> f16 bulk convert (for GEMM1 A operand)
__global__ void __launch_bounds__(256) cvt_f16_kernel(const float* __restrict__ src,
                                                      _Float16* __restrict__ dst) {
  size_t idx = (size_t)blockIdx.x * 256 + threadIdx.x;
  dst[idx] = (_Float16)src[idx];
}

// ---------------------------------------------------------------------------
// Row-major GEMM: C[M,N] = A[M,K] * Bpacked, f16 WMMA, f32 accumulate.
// One wave computes a 32x64 tile (2x4 subtiles). Block = 4 waves.
// Requires: M % 32 == 0, N % 64 == 0, K % 32 == 0. B is pre-packed (above).
// ---------------------------------------------------------------------------
template<typename TA, typename TC>
__global__ void __launch_bounds__(128) gemm_wmma(const TA* __restrict__ A,
                                                 const _Float16* __restrict__ Bp,
                                                 TC* __restrict__ C,
                                                 int N, int K, int lda, int ldc) {
  const int lane = threadIdx.x & 31;
  const int wave = threadIdx.x >> 5;
  const int stripsN = N >> 6;                    // N / 64
  const int strip = blockIdx.x * 4 + wave;
  const int sm = (strip / stripsN) << 5;         // * 32
  const int sn = (strip % stripsN) << 6;         // * 64
  const int mrow = lane & 15;
  const int hi = lane >> 4;                      // 0: lanes 0-15, 1: lanes 16-31
  const int kTiles = K >> 5;
  const v16h* Bv = (const v16h*)Bp;

  v8f acc[2][4];
  const v8f vz = {};
  #pragma unroll
  for (int mi = 0; mi < 2; mi++)
    #pragma unroll
    for (int ni = 0; ni < 4; ni++) acc[mi][ni] = vz;

  for (int k0 = 0; k0 < K; k0 += 32) {
    // A fragment (16x32 f16), contiguous k order, same as packed B:
    // lanes 0-15 : elems 0..15 -> K=k0+0..15
    // lanes 16-31: elems 0..15 -> K=k0+16..31
    v16h af[2];
    #pragma unroll
    for (int mi = 0; mi < 2; mi++) {
      const TA* ap = A + (size_t)(sm + mi * 16 + mrow) * lda + k0 + hi * 16;
      #pragma unroll
      for (int e = 0; e < 8; e++) af[mi][e] = (_Float16)ap[e];
      #pragma unroll
      for (int e = 0; e < 8; e++) af[mi][8 + e] = (_Float16)ap[8 + e];
    }
    // B fragment: one contiguous 32-byte vector load from packed layout
    v16h bf[4];
    #pragma unroll
    for (int ni = 0; ni < 4; ni++)
      bf[ni] = Bv[((size_t)((sn >> 4) + ni) * kTiles + (k0 >> 5)) * 32 + lane];

    #pragma unroll
    for (int mi = 0; mi < 2; mi++)
      #pragma unroll
      for (int ni = 0; ni < 4; ni++)
        acc[mi][ni] = __builtin_amdgcn_wmma_f32_16x16x32_f16(
            false, af[mi], false, bf[ni], (short)0, acc[mi][ni], false, false);
  }
  // C/D layout: VGPR i -> M = i (lanes 0-15) / i+8 (lanes 16-31); N = lane%16
  #pragma unroll
  for (int mi = 0; mi < 2; mi++)
    #pragma unroll
    for (int ni = 0; ni < 4; ni++)
      #pragma unroll
      for (int i = 0; i < 8; i++) {
        int row = sm + mi * 16 + i + hi * 8;
        int col = sn + ni * 16 + mrow;
        C[(size_t)row * ldc + col] = (TC)acc[mi][ni][i];
      }
}

// ---------------------------------------------------------------------------
// Depthwise causal conv (k=4) + SiLU for all 4 (stream,dir) combos.
// xz: [2][4096][2048] f16 (xi = cols 0..1023). dir=1 reads time-flipped.
// xf: [4][4096][1024] f16, combo c = stream*2 + dir, backward stored flipped.
// ---------------------------------------------------------------------------
__global__ void __launch_bounds__(256) conv_silu_kernel(
    const _Float16* __restrict__ xz,
    const float* __restrict__ wf, const float* __restrict__ cbf,
    const float* __restrict__ wb, const float* __restrict__ cbb,
    _Float16* __restrict__ xf) {
  size_t idx = (size_t)blockIdx.x * 256 + threadIdx.x;   // 4 * 4096 * 1024
  int d = idx & 1023;
  int r = (int)((idx >> 10) & 4095);
  int c = (int)(idx >> 22);
  int t = r & 1023, b = r >> 10;
  int dir = c & 1, s = c >> 1;
  const float* w  = dir ? wb  : wf;
  const float* cb = dir ? cbb : cbf;
  const _Float16* xs = xz + ((size_t)s << 23);
  float acc = cb[d];
  #pragma unroll
  for (int j = 0; j < 4; j++) {
    int ts = t - 3 + j;
    if (ts >= 0) {
      int tsrc = dir ? (1023 - ts) : ts;
      acc += w[d * 4 + j] * (float)xs[(((size_t)b << 10) + tsrc) * 2048 + d];
    }
  }
  float sl = acc / (1.f + __expf(-acc));                 // SiLU
  xf[idx] = (_Float16)sl;
}

// dt = softplus(dt_raw + b_dt[dir]), in place on f16 buffer [4][4096][1024]
__global__ void __launch_bounds__(256) dt_post_kernel(
    _Float16* __restrict__ dt,
    const float* __restrict__ bdf, const float* __restrict__ bdb) {
  size_t idx = (size_t)blockIdx.x * 256 + threadIdx.x;   // 16M
  int d = idx & 1023;
  int dir = (int)(idx >> 22) & 1;
  float x = (float)dt[idx] + (dir ? bdb : bdf)[d];
  float sp = (x > 20.f) ? x : log1pf(__expf(x));
  dt[idx] = (_Float16)sp;
}

// ---------------------------------------------------------------------------
// Selective scan: one lane per (combo, batch, channel) = 16384 lanes (512 waves).
// h[16] in VGPRs; u/dt prefetched 4 steps ahead; B/C are wave-broadcast, L0-hot.
// Backward combos write output time-flipped (un-flips the sequence).
// ---------------------------------------------------------------------------
__global__ void __launch_bounds__(256) scan_kernel(
    const _Float16* __restrict__ xf, const _Float16* __restrict__ dt,
    const float* __restrict__ dbc,
    const float* __restrict__ a_log_f, const float* __restrict__ a_log_b,
    const float* __restrict__ dsk_f,  const float* __restrict__ dsk_b,
    _Float16* __restrict__ y) {
  int tid = blockIdx.x * 256 + threadIdx.x;              // 0..16383
  int c = tid >> 12;
  int b = (tid >> 10) & 3;
  int d = tid & 1023;
  int dir = c & 1;
  const float* alog = dir ? a_log_b : a_log_f;
  float Dv = (dir ? dsk_b : dsk_f)[d];
  float Ac[16], h[16];
  #pragma unroll
  for (int n = 0; n < 16; n++) { Ac[n] = -__expf(alog[d * 16 + n]); h[n] = 0.f; }

  const size_t cb = ((size_t)c << 22) + ((size_t)b << 20) + d;
  const _Float16* xfp = xf + cb;
  const _Float16* dtp = dt + cb;
  _Float16* yp = y + cb;
  const float* bcp = dbc + ((size_t)c << 18) + ((size_t)b << 16);

  float ub[4], vb[4];
  #pragma unroll
  for (int q = 0; q < 4; q++) {
    ub[q] = (float)xfp[(size_t)q << 10];
    vb[q] = (float)dtp[(size_t)q << 10];
  }
  for (int t0 = 0; t0 < 1024; t0 += 4) {
    #pragma unroll
    for (int q = 0; q < 4; q++) {
      const int t = t0 + q;
      float u = ub[q], dtv = vb[q];
      if (t + 4 < 1024) {                                // refill pipeline
        ub[q] = (float)xfp[(size_t)(t + 4) << 10];
        vb[q] = (float)dtp[(size_t)(t + 4) << 10];
      }
      const float* bc = bcp + (size_t)t * 64 + 32;       // B = dbc[:,32:48], C = [48:64]
      float Bv[16], Cv[16];
      #pragma unroll
      for (int j = 0; j < 16; j++) { Bv[j] = bc[j]; Cv[j] = bc[16 + j]; }
      float du = dtv * u;
      float yv = 0.f;
      #pragma unroll
      for (int n = 0; n < 16; n++) {
        float dA = __expf(dtv * Ac[n]);
        h[n] = dA * h[n] + du * Bv[n];
        yv += h[n] * Cv[n];
      }
      yv += u * Dv;
      int to = dir ? (1023 - t) : t;
      yp[(size_t)to << 10] = (_Float16)yv;
    }
  }
}

// y = (yf + yb) * silu(z);  z = xz[:, 1024:2048]
__global__ void __launch_bounds__(256) combine_kernel(
    const _Float16* __restrict__ y, const _Float16* __restrict__ xz,
    _Float16* __restrict__ yc) {
  size_t idx = (size_t)blockIdx.x * 256 + threadIdx.x;   // 2 * 4096 * 1024
  int d = idx & 1023;
  int r = (int)((idx >> 10) & 4095);
  int s = (int)(idx >> 22);
  float z   = (float)xz[((size_t)s << 23) + (size_t)r * 2048 + 1024 + d];
  float yfv = (float)y[((size_t)(2 * s)     << 22) + ((size_t)r << 10) + d];
  float ybv = (float)y[((size_t)(2 * s + 1) << 22) + ((size_t)r << 10) + d];
  float sl = z / (1.f + __expf(-z));
  yc[idx] = (_Float16)((yfv + ybv) * sl);
}

// out = x + layernorm(yproj) ; one 128-thread block per 512-wide row
__global__ void __launch_bounds__(128) ln_res_kernel(
    const float* __restrict__ yproj,
    const float* __restrict__ ax, const float* __restrict__ vx,
    const float* __restrict__ g1, const float* __restrict__ b1,
    const float* __restrict__ g2, const float* __restrict__ b2,
    float* __restrict__ out) {
  int row = blockIdx.x;                                  // 0..8191
  int s = row >> 12;
  int r = row & 4095;
  const float* yrow = yproj + (size_t)row * 512;
  const float* xrow = (s ? vx : ax) + (size_t)r * 512;
  const float* g  = s ? g2 : g1;
  const float* bb = s ? b2 : b1;
  int tid = threadIdx.x;
  float v0 = yrow[tid], v1 = yrow[tid + 128], v2 = yrow[tid + 256], v3 = yrow[tid + 384];
  __shared__ float s1[128], s2[128];
  s1[tid] = v0 + v1 + v2 + v3;
  s2[tid] = v0 * v0 + v1 * v1 + v2 * v2 + v3 * v3;
  __syncthreads();
  for (int off = 64; off > 0; off >>= 1) {
    if (tid < off) { s1[tid] += s1[tid + off]; s2[tid] += s2[tid + off]; }
    __syncthreads();
  }
  float mu  = s1[0] * (1.f / 512.f);
  float var = s2[0] * (1.f / 512.f) - mu * mu;
  float rs = rsqrtf(var + 1e-6f);
  #pragma unroll
  for (int j = 0; j < 4; j++) {
    int i = tid + j * 128;
    out[(size_t)row * 512 + i] = xrow[i] + (yrow[i] - mu) * rs * g[i] + bb[i];
  }
}

// ---------------------------------------------------------------------------
extern "C" void kernel_launch(void* const* d_in, const int* in_sizes, int n_in,
                              void* d_out, int out_size, void* d_ws, size_t ws_size,
                              hipStream_t stream) {
  (void)in_sizes; (void)n_in; (void)out_size; (void)ws_size;
  const float* a_x      = (const float*)d_in[0];
  const float* v_x      = (const float*)d_in[1];
  const float* w_in     = (const float*)d_in[2];
  const float* conv_w_f = (const float*)d_in[3];
  const float* conv_b_f = (const float*)d_in[4];
  const float* conv_w_b = (const float*)d_in[5];
  const float* conv_b_b = (const float*)d_in[6];
  const float* w_x_f    = (const float*)d_in[7];
  const float* w_x_b    = (const float*)d_in[8];
  const float* w_dt_f   = (const float*)d_in[9];
  const float* b_dt_f   = (const float*)d_in[10];
  const float* w_dt_b   = (const float*)d_in[11];
  const float* b_dt_b   = (const float*)d_in[12];
  const float* a_log_f  = (const float*)d_in[13];
  const float* a_log_b  = (const float*)d_in[14];
  const float* d_skip_f = (const float*)d_in[15];
  const float* d_skip_b = (const float*)d_in[16];
  const float* w_out    = (const float*)d_in[17];
  const float* ln1_g    = (const float*)d_in[18];
  const float* ln1_b    = (const float*)d_in[19];
  const float* ln2_g    = (const float*)d_in[20];
  const float* ln2_b    = (const float*)d_in[21];
  float* out = (float*)d_out;

  // Workspace layout (~144 MB):
  char* ws = (char*)d_ws;
  _Float16* xz  = (_Float16*)(ws);                  // [2][4096][2048] f16 : 32 MB
  _Float16* xf  = (_Float16*)(ws + 33554432ull);    // [4][4096][1024] f16 : 32 MB
  float*    dbc = (float*)   (ws + 67108864ull);    // [4][4096][64]  f32 :  4 MB
  _Float16* dt  = (_Float16*)(ws + 71303168ull);    // [4][4096][1024] f16 : 32 MB
  _Float16* y   = (_Float16*)(ws + 104857600ull);   // [4][4096][1024] f16 : 32 MB
  _Float16* pw  = (_Float16*)(ws + 138412032ull);   // packed weights  : ~3.4 MB
  _Float16* pw_in   = pw;                           // 512x2048  -> 1048576 halves
  _Float16* pw_x_f  = pw + 1048576;                 // 1024x64   ->   65536
  _Float16* pw_x_b  = pw + 1114112;
  _Float16* pw_dt_f = pw + 1179648;                 // 32x1024   ->   32768
  _Float16* pw_dt_b = pw + 1212416;
  _Float16* pw_out  = pw + 1245184;                 // 1024x512  ->  524288
  _Float16* xh  = (_Float16*)(ws + 141950976ull);   // [2][4096][512] f16 : 8 MB
  _Float16* yc    = xf;                             // reuse xf after scans
  float*    yproj = (float*)(ws + 71303168ull);     // reuse dt region after scans

  // 0) pack weights into WMMA B-fragment layout; convert inputs to f16
  pack_b_kernel<<<256,256,0,stream>>>(w_in,   pw_in,   2048, 512, 2048);
  pack_b_kernel<<< 16,256,0,stream>>>(w_x_f,  pw_x_f,  64,   1024, 64);
  pack_b_kernel<<< 16,256,0,stream>>>(w_x_b,  pw_x_b,  64,   1024, 64);
  pack_b_kernel<<<  8,256,0,stream>>>(w_dt_f, pw_dt_f, 1024, 32,   1024);
  pack_b_kernel<<<  8,256,0,stream>>>(w_dt_b, pw_dt_b, 1024, 32,   1024);
  pack_b_kernel<<<128,256,0,stream>>>(w_out,  pw_out,  512,  1024, 512);
  cvt_f16_kernel<<<8192,256,0,stream>>>(a_x, xh);
  cvt_f16_kernel<<<8192,256,0,stream>>>(v_x, xh + (size_t)4096*512);

  // 1) in-projection: xz[s] = x_s @ w_in   (M=4096, N=2048, K=512)
  gemm_wmma<_Float16,_Float16><<<1024,128,0,stream>>>(xh,                    pw_in, xz,                     2048, 512, 512, 2048);
  gemm_wmma<_Float16,_Float16><<<1024,128,0,stream>>>(xh + (size_t)4096*512, pw_in, xz + (size_t)4096*2048, 2048, 512, 512, 2048);

  // 2) causal conv + SiLU, all 4 combos
  conv_silu_kernel<<<65536,256,0,stream>>>(xz, conv_w_f, conv_b_f, conv_w_b, conv_b_b, xf);

  // 3) x-projection: dbc[c] = xf[c] @ w_x_dir   (N=64, K=1024)
  for (int c = 0; c < 4; c++)
    gemm_wmma<_Float16,float><<<32,128,0,stream>>>(xf + (size_t)c*4194304,
        (c & 1) ? pw_x_b : pw_x_f, dbc + (size_t)c*262144, 64, 1024, 1024, 64);

  // 4) dt-projection: dt[c] = dbc[c][:, :32] @ w_dt_dir   (N=1024, K=32, lda=64)
  for (int c = 0; c < 4; c++)
    gemm_wmma<float,_Float16><<<512,128,0,stream>>>(dbc + (size_t)c*262144,
        (c & 1) ? pw_dt_b : pw_dt_f, dt + (size_t)c*4194304, 1024, 32, 64, 1024);

  // 5) dt = softplus(dt + b_dt)
  dt_post_kernel<<<65536,256,0,stream>>>(dt, b_dt_f, b_dt_b);

  // 6) all 4 selective scans concurrently (512 waves)
  scan_kernel<<<64,256,0,stream>>>(xf, dt, dbc, a_log_f, a_log_b, d_skip_f, d_skip_b, y);

  // 7) gated combine
  combine_kernel<<<32768,256,0,stream>>>(y, xz, yc);

  // 8) out-projection: yproj[s] = yc[s] @ w_out   (N=512, K=1024)
  for (int s = 0; s < 2; s++)
    gemm_wmma<_Float16,float><<<256,128,0,stream>>>(yc + (size_t)s*4194304, pw_out,
        yproj + (size_t)s*2097152, 512, 1024, 1024, 512);

  // 9) residual + layernorm -> d_out
  ln_res_kernel<<<8192,128,0,stream>>>(yproj, a_x, v_x, ln1_g, ln1_b, ln2_g, ln2_b, out);
}
